// RGCNModel2_13804024889642
// MI455X (gfx1250) — matrix operand order repaired and checked by reading, not compiled, not run on hardware
//
#include <hip/hip_runtime.h>
#include <hip/hip_fp16.h>
#include <math.h>

typedef __attribute__((ext_vector_type(16))) _Float16 v16h;
typedef __attribute__((ext_vector_type(8)))  float    v8f;

// Model constants (D=64, NUM_RELS=4, NUM_BASES=2). N_NODES divisible by 16.
#define DD 64

// ---------------------------------------------------------------------------
// Prep: W_m = sum_b w_comp[m,b]*basis[b] (m=0..3), W_4 = loop weight.
// Store directly in WMMA B-fragment layout (f16):
//   offset = m*4096 + kt*2048 + nt*512 + lane*16 + h
//   B-frag mapping: K = kt*32 + (lane>>4)*16 + h ; N = nt*16 + (lane&15)
// ---------------------------------------------------------------------------
__global__ __launch_bounds__(256)
void prep_weights_kernel(const float* __restrict__ basis,
                         const float* __restrict__ w_comp,
                         const float* __restrict__ loopw,
                         _Float16* __restrict__ bfrag) {
  int idx = blockIdx.x * blockDim.x + threadIdx.x;
  if (idx >= 5 * 4096) return;
  int h    = idx & 15;
  int lane = (idx >> 4) & 31;
  int nt   = (idx >> 9) & 3;
  int kt   = (idx >> 11) & 1;
  int m    = idx >> 12;           // 0..4 (4 = self-loop)
  int K = kt * 32 + (lane >> 4) * 16 + h;
  int n = nt * 16 + (lane & 15);
  float v;
  if (m < 4) {
    v = w_comp[m * 2 + 0] * basis[0 * 4096 + K * 64 + n]
      + w_comp[m * 2 + 1] * basis[1 * 4096 + K * 64 + n];
  } else {
    v = loopw[K * 64 + n];
  }
  bfrag[idx] = (_Float16)v;
}

// ---------------------------------------------------------------------------
// Projection: per 16-row tile (one wave), compute x_tile @ W_m for m=0..4.
// m=0..3 -> proj[N][4][64];  m=4 -> selfout[N][64] = x@loop + bias (seeds agg).
// A-frag mapping used here (same K order as bfrag): M = lane&15, K = hi*16 + h.
// D-frag: lane L, vgpr v -> C[M = v + 8*(L>>4)][N = L&15].
// ---------------------------------------------------------------------------
__global__ __launch_bounds__(128)
void proj_kernel(const float* __restrict__ x,
                 const _Float16* __restrict__ bfrag,
                 const float* __restrict__ bias,
                 float* __restrict__ proj,      // [N][4][64]
                 float* __restrict__ selfout,   // [N][64]
                 int ntiles, int do_relu) {
  int tile = blockIdx.x * 4 + (threadIdx.x >> 5);
  if (tile >= ntiles) return;                    // uniform per wave
  int lane = threadIdx.x & 31;
  int l  = lane & 15;
  int hi = lane >> 4;
  int row = tile * 16 + l;

  v16h a0, a1;
#pragma unroll
  for (int h = 0; h < 16; ++h) {
    int K = hi * 16 + h;
    float v0 = x[row * DD + K];
    float v1 = x[row * DD + 32 + K];
    if (do_relu) { v0 = fmaxf(v0, 0.f); v1 = fmaxf(v1, 0.f); }
    a0[h] = (_Float16)v0;
    a1[h] = (_Float16)v1;
  }

#pragma unroll
  for (int m = 0; m < 5; ++m) {
#pragma unroll
    for (int nt = 0; nt < 4; ++nt) {
      const v16h* b0p = (const v16h*)(bfrag + m * 4096 + 0 * 2048 + nt * 512 + lane * 16);
      const v16h* b1p = (const v16h*)(bfrag + m * 4096 + 1 * 2048 + nt * 512 + lane * 16);
      v16h b0 = *b0p;
      v16h b1 = *b1p;
      v8f acc = {};
      acc = __builtin_amdgcn_wmma_f32_16x16x32_f16(false, a0, false, b0, (short)0, acc, false, false);
      acc = __builtin_amdgcn_wmma_f32_16x16x32_f16(false, a1, false, b1, (short)0, acc, false, false);
      int ncol = nt * 16 + l;
      if (m < 4) {
#pragma unroll
        for (int v = 0; v < 8; ++v) {
          int r = tile * 16 + v + 8 * hi;
          proj[(r * 4 + m) * DD + ncol] = acc[v];
        }
      } else {
        float bv = bias[ncol];
#pragma unroll
        for (int v = 0; v < 8; ++v) {
          int r = tile * 16 + v + 8 * hi;
          selfout[r * DD + ncol] = acc[v] + bv;
        }
      }
    }
  }
}

// ---------------------------------------------------------------------------
// Edge scatter: 32 threads per edge, 2 cols each; gather proj[src, etype, :]
// and atomically accumulate into agg[dst, :] (native global_atomic_add_f32).
// proj (51 MB) + agg (12.8 MB) are L2-resident on MI455X (192 MB L2).
// ---------------------------------------------------------------------------
__global__ __launch_bounds__(256)
void edge_kernel(const int* __restrict__ src, const int* __restrict__ dst,
                 const int* __restrict__ et, const float* __restrict__ proj,
                 float* __restrict__ agg, int nE) {
  int i = blockIdx.x * blockDim.x + threadIdx.x;
  if (i >= nE * 32) return;
  int e = i >> 5;
  int c = (i & 31) * 2;
  int s = src[e], d = dst[e], t = et[e];
  const float2 v = *(const float2*)(proj + (s * 4 + t) * DD + c);
  unsafeAtomicAdd(agg + d * DD + c,     v.x);
  unsafeAtomicAdd(agg + d * DD + c + 1, v.y);
}

// ---------------------------------------------------------------------------
// Head: out[n] = sigmoid( relu(h2[n,:]) . pred_w + pred_b ).  One node/wave.
// ---------------------------------------------------------------------------
__global__ __launch_bounds__(256)
void head_kernel(const float* __restrict__ h, const float* __restrict__ pw,
                 const float* __restrict__ pb, float* __restrict__ out, int nN) {
  int node = blockIdx.x * 8 + (threadIdx.x >> 5);
  if (node >= nN) return;
  int lane = threadIdx.x & 31;
  float2 v = *(const float2*)(h + node * DD + lane * 2);
  float p = fmaxf(v.x, 0.f) * pw[lane * 2] + fmaxf(v.y, 0.f) * pw[lane * 2 + 1];
#pragma unroll
  for (int off = 16; off >= 1; off >>= 1)
    p += __shfl_xor(p, off, 32);
  if (lane == 0) out[node] = 1.f / (1.f + expf(-(p + pb[0])));
}

// ---------------------------------------------------------------------------
extern "C" void kernel_launch(void* const* d_in, const int* in_sizes, int n_in,
                              void* d_out, int out_size, void* d_ws, size_t ws_size,
                              hipStream_t stream) {
  const float* features = (const float*)d_in[0];
  const int*   src      = (const int*)d_in[1];
  const int*   dst      = (const int*)d_in[2];
  const int*   et       = (const int*)d_in[3];
  const float* basis1   = (const float*)d_in[4];
  const float* wcomp1   = (const float*)d_in[5];
  const float* loop1    = (const float*)d_in[6];
  const float* bias1    = (const float*)d_in[7];
  const float* basis2   = (const float*)d_in[8];
  const float* wcomp2   = (const float*)d_in[9];
  const float* loop2    = (const float*)d_in[10];
  const float* bias2    = (const float*)d_in[11];
  const float* predw    = (const float*)d_in[12];
  const float* predb    = (const float*)d_in[13];

  const int nN = in_sizes[0] / DD;        // 50000
  const int nE = in_sizes[1];             // 800000
  const int ntiles = nN / 16;             // 3125 (exact)

  // Workspace layout
  char* ws = (char*)d_ws;
  _Float16* bfrag1 = (_Float16*)(ws);                       //  40 KB
  _Float16* bfrag2 = (_Float16*)(ws + 40960);               //  40 KB
  float* proj = (float*)(ws + 81920);                       //  N*4*64*4 = 51.2 MB
  float* h1   = (float*)(ws + 81920 + (size_t)nN * 1024);   //  N*64*4   = 12.8 MB
  float* h2   = (float*)(ws + 81920 + (size_t)nN * 1024 + (size_t)nN * 256);

  prep_weights_kernel<<<80, 256, 0, stream>>>(basis1, wcomp1, loop1, bfrag1);
  prep_weights_kernel<<<80, 256, 0, stream>>>(basis2, wcomp2, loop2, bfrag2);

  const int pblocks = (ntiles + 3) / 4;
  const int eblocks = (nE * 32 + 255) / 256;
  const int hblocks = (nN + 7) / 8;

  // Layer 1
  proj_kernel<<<pblocks, 128, 0, stream>>>(features, bfrag1, bias1, proj, h1, ntiles, 0);
  edge_kernel<<<eblocks, 256, 0, stream>>>(src, dst, et, proj, h1, nE);
  // Layer 2 (ReLU fused into A-fragment load)
  proj_kernel<<<pblocks, 128, 0, stream>>>(h1, bfrag2, bias2, proj, h2, ntiles, 1);
  edge_kernel<<<eblocks, 256, 0, stream>>>(src, dst, et, proj, h2, nE);
  // Head (ReLU fused)
  head_kernel<<<hblocks, 256, 0, stream>>>(h2, predw, predb, (float*)d_out, nN);
}
